// TransformerEncoderLayerRotary_13984413515797
// MI455X (gfx1250) — matrix operand order repaired and checked by reading, not compiled, not run on hardware
//
#include <hip/hip_runtime.h>

// ---------------------------------------------------------------------------
// Transformer encoder layer w/ rotary embeddings for MI455X (gfx1250).
// GEMMs: v_wmma_f32_16x16x32_bf16 (wave32), LDS double-buffered staging via
// GLOBAL_LOAD_ASYNC_TO_LDS_B128 (ASYNCcnt), f32 accumulate.
// ---------------------------------------------------------------------------

typedef __bf16 bf16;
typedef __attribute__((ext_vector_type(16))) __bf16 v16bf;
typedef __attribute__((ext_vector_type(8)))  __bf16 v8bf;
typedef __attribute__((ext_vector_type(8)))  float  v8f;
typedef __attribute__((__vector_size__(16))) int    i32x4;

#define SEQ   2048
#define BATCH 2
#define DM    1024
#define NH    16
#define HD    64
#define DFF   4096
#define ROWS  (SEQ*BATCH)     // 4096 flattened (s,b) rows
#define BH    (BATCH*NH)      // 32 head-batches
#define LN_EPS 1e-5f

#define LDS_PITCH 40          // bf16 elems per LDS row (32 data + 8 pad = 80B)

// ---------------------------------------------------------------------------
// CDNA5 async global->LDS copy + wait (builtin if present, else inline asm)
// ---------------------------------------------------------------------------
__device__ __forceinline__ void async_b128(const bf16* g, bf16* l) {
#if __has_builtin(__builtin_amdgcn_global_load_async_to_lds_b128)
    typedef __attribute__((address_space(1))) i32x4 gvec_t;   // global int4
    typedef __attribute__((address_space(3))) i32x4 lvec_t;   // LDS int4
    __builtin_amdgcn_global_load_async_to_lds_b128(
        (gvec_t*)(uintptr_t)g,
        (lvec_t*)(unsigned)(uintptr_t)l,   // low 32 bits = LDS offset
        0, 0);
#else
    unsigned ldsoff = (unsigned)(uintptr_t)l;
    unsigned long long ga = (unsigned long long)(uintptr_t)g;
    asm volatile("global_load_async_to_lds_b128 %0, %1, off"
                 :: "v"(ldsoff), "v"(ga) : "memory");
#endif
}

__device__ __forceinline__ void wait_async_le3() {
#if __has_builtin(__builtin_amdgcn_s_wait_asynccnt)
    __builtin_amdgcn_s_wait_asynccnt(3);
#else
    asm volatile("s_wait_asynccnt 3" ::: "memory");
#endif
}
__device__ __forceinline__ void wait_async_le0() {
#if __has_builtin(__builtin_amdgcn_s_wait_asynccnt)
    __builtin_amdgcn_s_wait_asynccnt(0);
#else
    asm volatile("s_wait_asynccnt 0" ::: "memory");
#endif
}

// ---------------------------------------------------------------------------
// WMMA helpers
// ---------------------------------------------------------------------------
__device__ __forceinline__ v8f wmma_bf16(v16bf a, v16bf b, v8f c) {
    return __builtin_amdgcn_wmma_f32_16x16x32_bf16(
        /*neg_a=*/false, a, /*neg_b=*/false, b,
        /*c_mod=*/(short)0, c, /*reuse_a=*/false, /*reuse_b=*/false);
}

// A operand 16x32 bf16 from an LDS tile (row pitch LDS_PITCH).
// Lanes 0-15: row=lane, K {0..7} then {16..23}; lanes 16-31: {8..15},{24..31}
__device__ __forceinline__ v16bf lds_frag_a(const bf16* base, int row, int half) {
    const bf16* p = base + row * LDS_PITCH + (half << 3);
    v8bf lo = *reinterpret_cast<const v8bf*>(p);        // 16B, ds_load_b128
    v8bf hi = *reinterpret_cast<const v8bf*>(p + 16);
    v16bf f;
#pragma unroll
    for (int i = 0; i < 8; ++i) { f[i] = lo[i]; f[i + 8] = hi[i]; }
    return f;
}

// B operand 32x16 bf16 from an LDS tile of Bt rows (row n = column of B).
// Lanes 0-15: col=lane, K {0..7} then {16..23}; lanes 16-31: {8..15},{24..31}
__device__ __forceinline__ v16bf lds_frag_b(const bf16* base, int row, int half) {
    const bf16* p = base + row * LDS_PITCH + (half << 3);
    v8bf lo = *reinterpret_cast<const v8bf*>(p);
    v8bf hi = *reinterpret_cast<const v8bf*>(p + 16);
    v16bf f;
#pragma unroll
    for (int i = 0; i < 8; ++i) { f[i] = lo[i]; f[i + 8] = hi[i]; }
    return f;
}

// ---------------------------------------------------------------------------
// Generic WMMA GEMM: C[M,N] = alpha * (A[M,K] @ Bt[N,K]^T) + bias[N]
// Block tile 128x64x32, 8 waves (4x2), each wave 32x32 (2x2 WMMA frags).
// A/B K-slabs staged in LDS with double-buffered async copies.
// grid = (M/128, N/64). Optional ReLU; f32 and/or bf16 output.
// ---------------------------------------------------------------------------
__global__ __launch_bounds__(256)
void gemm_bf16_wmma(const bf16* __restrict__ A, int lda,
                    const bf16* __restrict__ Bt, int ldb,
                    const float* __restrict__ bias,
                    float* __restrict__ Cf, bf16* __restrict__ Cb, int ldc,
                    int K, float alpha, int relu) {
    __shared__ bf16 As[2][128 * LDS_PITCH];   // 2 x 10240B
    __shared__ bf16 Bs[2][64 * LDS_PITCH];    // 2 x  5120B

    const int tid  = threadIdx.x;
    const int lane = tid & 31;
    const int wave = tid >> 5;                 // 0..7
    const int r    = lane & 15;
    const int half = lane >> 4;
    const int mw   = (wave >> 1) * 32;         // wave tile offset within block
    const int nw   = (wave & 1) * 32;
    const long blockM = (long)blockIdx.x * 128;
    const long blockN = (long)blockIdx.y * 64;

    // Per-thread async copy slots: A = 512 x 16B chunks (2/thread), B = 256 x 16B.
    const int ar0 = tid >> 2,          ac0 = (tid & 3) * 8;           // A chunk 0
    const int ar1 = (tid + 256) >> 2,  ac1 = ac0;                     // A chunk 1
    const int br  = tid >> 2,          bc  = (tid & 3) * 8;           // B chunk

    auto issue_stage = [&](int buf, int k0) {
        async_b128(A  + (blockM + ar0) * lda + k0 + ac0, &As[buf][ar0 * LDS_PITCH + ac0]);
        async_b128(A  + (blockM + ar1) * lda + k0 + ac1, &As[buf][ar1 * LDS_PITCH + ac1]);
        async_b128(Bt + (blockN + br ) * ldb + k0 + bc , &Bs[buf][br  * LDS_PITCH + bc ]);
    };

    v8f acc00 = {}, acc01 = {}, acc10 = {}, acc11 = {};

    issue_stage(0, 0);
    int buf = 0;
    for (int k0 = 0; k0 < K; k0 += 32) {
        const bool more = (k0 + 32) < K;
        if (more) {                       // prefetch next slab on the async engine
            issue_stage(buf ^ 1, k0 + 32);
            wait_async_le3();             // in-order: current slab's 3 copies done
        } else {
            wait_async_le0();
        }
        __syncthreads();                  // slab visible to all waves

        const bf16* Ab = As[buf];
        const bf16* Bb = Bs[buf];
        v16bf a0 = lds_frag_a(Ab, mw + r,      half);
        v16bf a1 = lds_frag_a(Ab, mw + 16 + r, half);
        v16bf b0 = lds_frag_b(Bb, nw + r,      half);
        v16bf b1 = lds_frag_b(Bb, nw + 16 + r, half);
        acc00 = wmma_bf16(a0, b0, acc00);
        acc01 = wmma_bf16(a0, b1, acc01);
        acc10 = wmma_bf16(a1, b0, acc10);
        acc11 = wmma_bf16(a1, b1, acc11);

        __syncthreads();                  // all waves done reading before overwrite
        buf ^= 1;
    }

    auto store_tile = [&](v8f acc, long mb, long nb) {
#pragma unroll
        for (int i = 0; i < 8; ++i) {
            long row = mb + half * 8 + i;   // C/D layout: lanes 0-15 M=i, 16-31 M=8+i
            long col = nb + r;
            float v = acc[i] * alpha + (bias ? bias[col] : 0.0f);
            if (relu) v = fmaxf(v, 0.0f);
            if (Cf) Cf[row * ldc + col] = v;
            if (Cb) Cb[row * ldc + col] = (bf16)v;
        }
    };
    store_tile(acc00, blockM + mw,      blockN + nw);
    store_tile(acc01, blockM + mw,      blockN + nw + 16);
    store_tile(acc10, blockM + mw + 16, blockN + nw);
    store_tile(acc11, blockM + mw + 16, blockN + nw + 16);
}

// ---------------------------------------------------------------------------
// Elementwise f32 -> bf16
// ---------------------------------------------------------------------------
__global__ __launch_bounds__(256)
void cvt_bf16_kernel(const float* __restrict__ in, bf16* __restrict__ out, long n) {
    long i = (long)blockIdx.x * 256 + threadIdx.x;
    if (i < n) out[i] = (bf16)in[i];
}

// Transpose + convert: Wt[n,k] = (bf16)W[k,n].  W is [K,N]. grid=(K/32, N/32)
__global__ __launch_bounds__(256)
void transpose_cvt_kernel(const float* __restrict__ W, bf16* __restrict__ Wt,
                          int K, int N) {
    __shared__ float tile[32][33];
    int k0 = blockIdx.x * 32, n0 = blockIdx.y * 32;
    int tx = threadIdx.x & 31, ty = threadIdx.x >> 5;   // ty 0..7
#pragma unroll
    for (int i = ty; i < 32; i += 8)
        tile[i][tx] = W[(long)(k0 + i) * N + (n0 + tx)];
    __syncthreads();
#pragma unroll
    for (int i = ty; i < 32; i += 8)
        Wt[(long)(n0 + i) * K + (k0 + tx)] = (bf16)tile[tx][i];
}

// ---------------------------------------------------------------------------
// RoPE (interleaved pairs) + repack [S,B,E] f32 -> [B,H,S,64] bf16
// ---------------------------------------------------------------------------
__global__ __launch_bounds__(256)
void rope_pack_kernel(const float* __restrict__ X, bf16* __restrict__ O) {
    long idx = (long)blockIdx.x * 256 + threadIdx.x;    // over B*H*S*(HD/2) = 2^21
    int d2 = (int)(idx & 31);
    int s  = (int)((idx >> 5) & (SEQ - 1));
    int h  = (int)((idx >> 16) & (NH - 1));
    int b  = (int)(idx >> 20);
    const float* p = X + (long)s * (BATCH * DM) + b * DM + h * HD + d2 * 2;
    float x0 = p[0], x1 = p[1];
    // inv_freq = theta^(-2*d2/64); theta=10000 -> exp(-(2*d2/64)*ln(10000))
    float inv = __expf(-((float)(2 * d2) / (float)HD) * 9.210340371976184f);
    float f = (float)s * inv;
    float c = __cosf(f), sn = __sinf(f);
    bf16* o = O + ((long)(b * NH + h) * SEQ + s) * HD + d2 * 2;
    o[0] = (bf16)(x0 * c - x1 * sn);   // pair rotation (x0,x1)
    o[1] = (bf16)(x1 * c + x0 * sn);
}

// V: [S,B,E] f32 -> Vt [B,H,64,S] bf16 (so PV GEMM's B operand is [N=d, K=s])
__global__ __launch_bounds__(256)
void vpack_kernel(const float* __restrict__ X, bf16* __restrict__ O) {
    long idx = (long)blockIdx.x * 256 + threadIdx.x;    // over B*H*HD*S = 2^22
    int s = (int)(idx & (SEQ - 1));
    int d = (int)((idx >> 11) & (HD - 1));
    int h = (int)((idx >> 17) & (NH - 1));
    int b = (int)(idx >> 21);
    O[idx] = (bf16)X[(long)s * (BATCH * DM) + b * DM + h * HD + d];
}

// ---------------------------------------------------------------------------
// Row softmax over 2048 columns -> bf16 P.  grid = 2048 rows, 256 threads.
// ---------------------------------------------------------------------------
__global__ __launch_bounds__(256)
void softmax_kernel(const float* __restrict__ Sc, bf16* __restrict__ P) {
    __shared__ float red[256];
    int row = blockIdx.x, tid = threadIdx.x;
    const float* rp = Sc + (long)row * SEQ;
    float v[8];
    float m = -3.0e38f;
#pragma unroll
    for (int i = 0; i < 8; ++i) { v[i] = rp[tid + i * 256]; m = fmaxf(m, v[i]); }
    red[tid] = m; __syncthreads();
    for (int off = 128; off > 0; off >>= 1) {
        if (tid < off) red[tid] = fmaxf(red[tid], red[tid + off]);
        __syncthreads();
    }
    m = red[0]; __syncthreads();
    float sum = 0.f;
#pragma unroll
    for (int i = 0; i < 8; ++i) { v[i] = __expf(v[i] - m); sum += v[i]; }
    red[tid] = sum; __syncthreads();
    for (int off = 128; off > 0; off >>= 1) {
        if (tid < off) red[tid] += red[tid + off];
        __syncthreads();
    }
    float inv = 1.0f / red[0];
    bf16* pp = P + (long)row * SEQ;
#pragma unroll
    for (int i = 0; i < 8; ++i) pp[tid + i * 256] = (bf16)(v[i] * inv);
}

// ---------------------------------------------------------------------------
// out = LayerNorm(A + Bres) * gamma + beta ; optional f32 + bf16 outputs.
// One block per 1024-wide row; grid = 4096.
// ---------------------------------------------------------------------------
__global__ __launch_bounds__(256)
void add_ln_kernel(const float* __restrict__ A, const float* __restrict__ Bres,
                   const float* __restrict__ gamma, const float* __restrict__ beta,
                   float* __restrict__ outF, bf16* __restrict__ outB) {
    __shared__ float red[256];
    int row = blockIdx.x, tid = threadIdx.x;
    const float* pa = A + (long)row * DM;
    const float* pb = Bres + (long)row * DM;
    float v[4];
    float s = 0.f;
#pragma unroll
    for (int i = 0; i < 4; ++i) { v[i] = pa[tid + i * 256] + pb[tid + i * 256]; s += v[i]; }
    red[tid] = s; __syncthreads();
    for (int off = 128; off > 0; off >>= 1) {
        if (tid < off) red[tid] += red[tid + off];
        __syncthreads();
    }
    float mu = red[0] * (1.0f / DM); __syncthreads();
    float s2 = 0.f;
#pragma unroll
    for (int i = 0; i < 4; ++i) { float d = v[i] - mu; s2 += d * d; }
    red[tid] = s2; __syncthreads();
    for (int off = 128; off > 0; off >>= 1) {
        if (tid < off) red[tid] += red[tid + off];
        __syncthreads();
    }
    float rstd = rsqrtf(red[0] * (1.0f / DM) + LN_EPS);
#pragma unroll
    for (int i = 0; i < 4; ++i) {
        int c = tid + i * 256;
        float o = (v[i] - mu) * rstd * gamma[c] + beta[c];
        if (outF) outF[(long)row * DM + c] = o;
        if (outB) outB[(long)row * DM + c] = (bf16)o;
    }
}

// ---------------------------------------------------------------------------
// Host orchestration
// ---------------------------------------------------------------------------
extern "C" void kernel_launch(void* const* d_in, const int* in_sizes, int n_in,
                              void* d_out, int out_size, void* d_ws, size_t ws_size,
                              hipStream_t stream) {
    (void)in_sizes; (void)n_in; (void)out_size; (void)ws_size;
    const float* src   = (const float*)d_in[0];
    const float* q_w   = (const float*)d_in[1];
    const float* q_b   = (const float*)d_in[2];
    const float* k_w   = (const float*)d_in[3];
    const float* k_b   = (const float*)d_in[4];
    const float* v_w   = (const float*)d_in[5];
    const float* v_b   = (const float*)d_in[6];
    const float* w1    = (const float*)d_in[7];
    const float* b1    = (const float*)d_in[8];
    const float* w2    = (const float*)d_in[9];
    const float* b2    = (const float*)d_in[10];
    const float* ln1_g = (const float*)d_in[11];
    const float* ln1_b = (const float*)d_in[12];
    const float* ln2_g = (const float*)d_in[13];
    const float* ln2_b = (const float*)d_in[14];
    float* out = (float*)d_out;

    char* w = (char*)d_ws;
    size_t off = 0;
    auto alloc = [&](size_t bytes) -> char* {
        char* p = w + off;
        off += (bytes + 255) & ~(size_t)255;
        return p;
    };
    bf16*  srcB   = (bf16*)alloc((size_t)ROWS * DM * 2);
    bf16*  qWt    = (bf16*)alloc((size_t)DM * DM * 2);
    bf16*  kWt    = (bf16*)alloc((size_t)DM * DM * 2);
    bf16*  vWt    = (bf16*)alloc((size_t)DM * DM * 2);
    bf16*  w1t    = (bf16*)alloc((size_t)DFF * DM * 2);
    bf16*  w2t    = (bf16*)alloc((size_t)DM * DFF * 2);
    float* qf     = (float*)alloc((size_t)ROWS * DM * 4);
    float* kf     = (float*)alloc((size_t)ROWS * DM * 4);
    float* vf     = (float*)alloc((size_t)ROWS * DM * 4);
    bf16*  qbh    = (bf16*)alloc((size_t)BH * SEQ * HD * 2);
    bf16*  kbh    = (bf16*)alloc((size_t)BH * SEQ * HD * 2);
    bf16*  vT     = (bf16*)alloc((size_t)BH * HD * SEQ * 2);
    float* scores = (float*)alloc((size_t)SEQ * SEQ * 4);
    bf16*  P      = (bf16*)alloc((size_t)SEQ * SEQ * 2);
    float* attn   = (float*)alloc((size_t)ROWS * DM * 4);
    float* xf     = (float*)alloc((size_t)ROWS * DM * 4);
    bf16*  xb     = (bf16*)alloc((size_t)ROWS * DM * 2);
    bf16*  hb     = (bf16*)alloc((size_t)ROWS * DFF * 2);
    float* ff     = (float*)alloc((size_t)ROWS * DM * 4);

    // ---- precision conversion / weight transposes ----
    cvt_bf16_kernel<<<(ROWS * DM) / 256, 256, 0, stream>>>(src, srcB, (long)ROWS * DM);
    transpose_cvt_kernel<<<dim3(DM / 32, DM / 32), 256, 0, stream>>>(q_w, qWt, DM, DM);
    transpose_cvt_kernel<<<dim3(DM / 32, DM / 32), 256, 0, stream>>>(k_w, kWt, DM, DM);
    transpose_cvt_kernel<<<dim3(DM / 32, DM / 32), 256, 0, stream>>>(v_w, vWt, DM, DM);
    transpose_cvt_kernel<<<dim3(DM / 32, DFF / 32), 256, 0, stream>>>(w1, w1t, DM, DFF);
    transpose_cvt_kernel<<<dim3(DFF / 32, DM / 32), 256, 0, stream>>>(w2, w2t, DFF, DM);

    // ---- QKV projections (4096x1024x1024, WMMA bf16) ----
    dim3 gQKV(ROWS / 128, DM / 64);
    gemm_bf16_wmma<<<gQKV, 256, 0, stream>>>(srcB, DM, qWt, DM, q_b, qf, nullptr, DM, DM, 1.0f, 0);
    gemm_bf16_wmma<<<gQKV, 256, 0, stream>>>(srcB, DM, kWt, DM, k_b, kf, nullptr, DM, DM, 1.0f, 0);
    gemm_bf16_wmma<<<gQKV, 256, 0, stream>>>(srcB, DM, vWt, DM, v_b, vf, nullptr, DM, DM, 1.0f, 0);

    // ---- RoPE + head-major repack ----
    rope_pack_kernel<<<(BH * SEQ * (HD / 2)) / 256, 256, 0, stream>>>(qf, qbh);
    rope_pack_kernel<<<(BH * SEQ * (HD / 2)) / 256, 256, 0, stream>>>(kf, kbh);
    vpack_kernel<<<(BH * SEQ * HD) / 256, 256, 0, stream>>>(vf, vT);

    // ---- attention, head-serial (stream order lets us reuse scores/P) ----
    const float isqrt_hd = 0.125f;   // 1/sqrt(64)
    for (int bh = 0; bh < BH; ++bh) {
        const bf16* qh = qbh + (long)bh * SEQ * HD;
        const bf16* kh = kbh + (long)bh * SEQ * HD;
        const bf16* vh = vT  + (long)bh * HD * SEQ;
        int b = bh >> 4, h = bh & (NH - 1);
        // scores = (q @ k^T) / sqrt(hd) : M=N=2048, K=64
        gemm_bf16_wmma<<<dim3(SEQ / 128, SEQ / 64), 256, 0, stream>>>(
            qh, HD, kh, HD, nullptr, scores, nullptr, SEQ, HD, isqrt_hd, 0);
        softmax_kernel<<<SEQ, 256, 0, stream>>>(scores, P);
        // out_head = P @ V : M=2048, N=64, K=2048, scattered into attn[S,B,E]
        gemm_bf16_wmma<<<dim3(SEQ / 128, HD / 64), 256, 0, stream>>>(
            P, SEQ, vh, SEQ, nullptr, attn + (long)b * DM + h * HD, nullptr,
            BATCH * DM, SEQ, 1.0f, 0);
    }

    // ---- x = LN(src + attn) ----
    add_ln_kernel<<<ROWS, 256, 0, stream>>>(attn, src, ln1_g, ln1_b, xf, xb);

    // ---- FFN: h = relu(x@w1+b1) (bf16), ff = h@w2+b2 (f32) ----
    gemm_bf16_wmma<<<dim3(ROWS / 128, DFF / 64), 256, 0, stream>>>(
        xb, DM, w1t, DM, b1, nullptr, hb, DFF, DM, 1.0f, 1);
    gemm_bf16_wmma<<<dim3(ROWS / 128, DM / 64), 256, 0, stream>>>(
        hb, DFF, w2t, DFF, b2, ff, nullptr, DM, DFF, 1.0f, 0);

    // ---- out = LN(x + ff) ----
    add_ln_kernel<<<ROWS, 256, 0, stream>>>(ff, xf, ln2_g, ln2_b, out, nullptr);
}
